// Net_37469294690489
// MI455X (gfx1250) — compile-verified, matrix operand order checked by reading, not run on hardware
//
#include <hip/hip_runtime.h>
#include <math.h>

// ---------------------------------------------------------------------------
// Problem constants (from reference)
// ---------------------------------------------------------------------------
#define NPROT 256
#define NGOS  20000
#define HID   1024
#define NCLS  5000
#define EGO   10240000
#define EBLK  8192
#define NNODE (NPROT * NGOS) // 5,120,000

typedef __attribute__((ext_vector_type(16))) __bf16 v16bf;
typedef __attribute__((ext_vector_type(8)))  __bf16 v8bf;
typedef __attribute__((ext_vector_type(2)))  __bf16 v2bf;
typedef __attribute__((ext_vector_type(8)))  float  v8f;
typedef __attribute__((ext_vector_type(4)))  float  v4f;

// LDS row stride (bf16 elems) for B tiles: 40 elems = 80 B keeps ds_load_b128
// 16B-aligned and spreads rows across banks.
#define LDSTR 40

// ---------------------------------------------------------------------------
// CDNA5 async global->LDS DMA (VGLOBAL GLOBAL_LOAD_ASYNC_TO_LDS_B128, GV mode)
// Tracked by ASYNCcnt; completion enforced with s_wait_asynccnt.
// ---------------------------------------------------------------------------
__device__ __forceinline__ void async_copy16(void* lds_dst, const void* gsrc) {
    unsigned lo = (unsigned)(size_t)lds_dst;   // low 32 bits = LDS byte offset
    asm volatile("global_load_async_to_lds_b128 %0, %1, off"
                 :: "v"(lo), "v"(gsrc) : "memory");
}
__device__ __forceinline__ void wait_async0() {
    asm volatile("s_wait_asynccnt 0" ::: "memory");
}

// ---------------------------------------------------------------------------
// Utility kernels
// ---------------------------------------------------------------------------
__global__ void zero_f32(float* __restrict__ p, int n) {
    int i = blockIdx.x * blockDim.x + threadIdx.x;
    if (i < n) p[i] = 0.0f;
}

// RelGraphConv message scatter: agg[dst] += h[src] * w_rel[etype]
// h is x.T.reshape(N): h[i] = x[(i&255)*NGOS + (i>>8)]
__global__ __launch_bounds__(256)
void rel_scatter(const int* __restrict__ src, const int* __restrict__ dst,
                 const int* __restrict__ et,  const float* __restrict__ x,
                 const float* __restrict__ w_rel, float* __restrict__ agg) {
    int e = blockIdx.x * blockDim.x + threadIdx.x;
    if (e >= EGO) return;
    int s = src[e];
    int p = s & (NPROT - 1);
    int g = s >> 8;
    float hval = x[(size_t)p * NGOS + g];
    float msg  = hval * w_rel[et[e]];
    atomicAdd(&agg[dst[e]], msg);
}

__global__ void blk_degrees(const int* __restrict__ bsrc, const int* __restrict__ bdst,
                            float* __restrict__ degout, float* __restrict__ degin) {
    int e = blockIdx.x * blockDim.x + threadIdx.x;
    if (e >= EBLK) return;
    atomicAdd(&degout[bsrc[e]], 1.0f);
    atomicAdd(&degin[bdst[e]], 1.0f);
}

// m[dst,:] += H1[src,:] * deg_out[src]^-1/2   (one block per edge)
__global__ __launch_bounds__(256)
void blk_scatter(const int* __restrict__ bsrc, const int* __restrict__ bdst,
                 const float* __restrict__ H1, const float* __restrict__ degout,
                 float* __restrict__ m) {
    int e = blockIdx.x;
    int s = bsrc[e], d = bdst[e];
    float scale = rsqrtf(fmaxf(degout[s], 1.0f));
    for (int c = threadIdx.x; c < HID; c += blockDim.x)
        atomicAdd(&m[(size_t)d * HID + c], H1[(size_t)s * HID + c] * scale);
}

__global__ void blk_norm_in(float* __restrict__ m, const float* __restrict__ degin) {
    int i = blockIdx.x * blockDim.x + threadIdx.x;
    if (i >= NPROT * HID) return;
    int p = i >> 10;
    m[i] *= rsqrtf(fmaxf(degin[p], 1.0f));
}

// ---------------------------------------------------------------------------
// Fragment helpers.  Per ISA 7.12.2 the 16-bit A/B fragment of a lane is two
// contiguous runs of 8 K-values: [half*8, +8) and [16+half*8, +8).
// ---------------------------------------------------------------------------
// bf16 tile in LDS, K-contiguous rows -> two ds_load_b128
__device__ __forceinline__ v16bf frag_bf16(const __bf16* row, int half) {
    v8bf lo = *(const v8bf*)(row + half * 8);
    v8bf hi = *(const v8bf*)(row + 16 + half * 8);
    return __builtin_shufflevector(lo, hi, 0, 1, 2, 3, 4, 5, 6, 7,
                                   8, 9, 10, 11, 12, 13, 14, 15);
}
// fp32 tile in LDS (async-DMA'd, row stride 32 floats) -> 4x ds_load_b128 + cvt
__device__ __forceinline__ v16bf frag_f32(const float* row, int half) {
    v4f r0 = *(const v4f*)(row + half * 8);
    v4f r1 = *(const v4f*)(row + half * 8 + 4);
    v4f r2 = *(const v4f*)(row + 16 + half * 8);
    v4f r3 = *(const v4f*)(row + 16 + half * 8 + 4);
    v16bf a;
#pragma unroll
    for (int j = 0; j < 4; ++j) {
        a[j]      = (__bf16)r0[j];
        a[j + 4]  = (__bf16)r1[j];
        a[j + 8]  = (__bf16)r2[j];
        a[j + 12] = (__bf16)r3[j];
    }
    return a;
}

// ---------------------------------------------------------------------------
// GEMM 1: H1[256,1024] = relu( A0 @ fc0_w + fc0_b )
// A0[p][k] fused on the fly: agg[k*256+p] + x[p*NGOS+k]*w_loop + b_rel
// Block = 128 threads (4 waves) -> 16(M) x 64(N) tile; K stepped by 32.
// Double-buffered LDS: stage tile k+1 while WMMA consumes tile k.
// ---------------------------------------------------------------------------
__global__ __launch_bounds__(128)
void fc0_gemm(const float* __restrict__ agg, const float* __restrict__ x,
              const float* __restrict__ wloop, const float* __restrict__ brel,
              const float* __restrict__ W, const float* __restrict__ bias,
              float* __restrict__ out) {
    __shared__ __align__(16) __bf16 Ash[2][16 * LDSTR];  // [m][k]
    __shared__ __align__(16) __bf16 Bsh[2][64 * LDSTR];  // [n][k] transposed

    int tid  = threadIdx.x;
    int wv   = tid >> 5;
    int lane = tid & 31;
    int mn = lane & 15, half = lane >> 4;
    int m0 = (blockIdx.x >> 4) * 16;
    int n0 = (blockIdx.x & 15) * 64;
    const float wl = wloop[0], br = brel[0];

    auto fill = [&](int buf, int k0) {
        // A: 16x32 fp32 -> bf16, fused RelGraphConv epilogue
        for (int i = tid; i < 256; i += 128) {
            int m  = i >> 4;
            int kk = (i & 15) * 2;
            int p  = m0 + m;
            int k  = k0 + kk;
            float a0 = agg[(size_t)k * NPROT + p]       + x[(size_t)p * NGOS + k]     * wl + br;
            float a1 = agg[(size_t)(k + 1) * NPROT + p] + x[(size_t)p * NGOS + k + 1] * wl + br;
            *(v2bf*)&Ash[buf][m * LDSTR + kk] = (v2bf){(__bf16)a0, (__bf16)a1};
        }
        // B: 32x64 fp32, stored K-transposed [n][k] as bf16
        for (int i = tid; i < 1024; i += 128) {
            int c = i & 63;
            int r = (i >> 6) * 2;
            float b0 = W[(size_t)(k0 + r) * HID + n0 + c];
            float b1 = W[(size_t)(k0 + r + 1) * HID + n0 + c];
            *(v2bf*)&Bsh[buf][c * LDSTR + r] = (v2bf){(__bf16)b0, (__bf16)b1};
        }
    };

    fill(0, 0);
    __syncthreads();
    v8f acc = {};
    for (int k0 = 0; k0 < NGOS; k0 += 32) {
        int cur = (k0 >> 5) & 1, nxt = cur ^ 1;
        v16bf a = frag_bf16(&Ash[cur][mn * LDSTR], half);
        v16bf b = frag_bf16(&Bsh[cur][(wv * 16 + mn) * LDSTR], half);
        if (k0 + 32 < NGOS) {
            __builtin_prefetch(&W[(size_t)(k0 + 64) * HID + n0 + lane * 2], 0, 1);
            fill(nxt, k0 + 32);
        }
        acc = __builtin_amdgcn_wmma_f32_16x16x32_bf16(false, a, false, b,
                                                      (short)0, acc, false, false);
        __syncthreads();
    }
#pragma unroll
    for (int r = 0; r < 8; ++r) {
        int row = m0 + half * 8 + r;
        int col = n0 + wv * 16 + mn;
        float v = acc[r] + bias[col];
        out[(size_t)row * HID + col] = v > 0.0f ? v : 0.0f;
    }
}

// ---------------------------------------------------------------------------
// GEMM 2: H2[256,1024] = mn @ gcn_w + gcn_b     (K = 1024)
// A tile (pure fp32 row block) DMA'd with global_load_async_to_lds_b128.
// ---------------------------------------------------------------------------
__global__ __launch_bounds__(128)
void gcn_gemm(const float* __restrict__ A, const float* __restrict__ W,
              const float* __restrict__ bias, float* __restrict__ out) {
    __shared__ __align__(16) float  Afp[2][16 * 32];     // [m][k] fp32
    __shared__ __align__(16) __bf16 Bsh[2][64 * LDSTR];  // [n][k] bf16

    int tid  = threadIdx.x;
    int wv   = tid >> 5;
    int lane = tid & 31;
    int mn = lane & 15, half = lane >> 4;
    int m0 = (blockIdx.x >> 4) * 16;
    int n0 = (blockIdx.x & 15) * 64;

    auto fill = [&](int buf, int k0) {
        // A: one async 16B DMA per thread (16 rows x 128 B)
        int row = tid >> 3;
        int seg = (tid & 7) * 4;
        async_copy16(&Afp[buf][row * 32 + seg],
                     &A[(size_t)(m0 + row) * HID + k0 + seg]);
        // B: cooperative convert + transpose
        for (int i = tid; i < 1024; i += 128) {
            int c = i & 63;
            int r = (i >> 6) * 2;
            float b0 = W[(size_t)(k0 + r) * HID + n0 + c];
            float b1 = W[(size_t)(k0 + r + 1) * HID + n0 + c];
            *(v2bf*)&Bsh[buf][c * LDSTR + r] = (v2bf){(__bf16)b0, (__bf16)b1};
        }
    };

    fill(0, 0);
    wait_async0();
    __syncthreads();
    v8f acc = {};
    for (int k0 = 0; k0 < HID; k0 += 32) {
        int cur = (k0 >> 5) & 1, nxt = cur ^ 1;
        v16bf a = frag_f32(&Afp[cur][mn * 32], half);
        v16bf b = frag_bf16(&Bsh[cur][(wv * 16 + mn) * LDSTR], half);
        if (k0 + 32 < HID) {
            __builtin_prefetch(&W[(size_t)(k0 + 64) * HID + n0 + lane * 2], 0, 1);
            fill(nxt, k0 + 32);
        }
        acc = __builtin_amdgcn_wmma_f32_16x16x32_bf16(false, a, false, b,
                                                      (short)0, acc, false, false);
        wait_async0();
        __syncthreads();
    }
#pragma unroll
    for (int r = 0; r < 8; ++r) {
        int row = m0 + half * 8 + r;
        int col = n0 + wv * 16 + mn;
        out[(size_t)row * HID + col] = acc[r] + bias[col];
    }
}

// ---------------------------------------------------------------------------
// GEMM 3: OUT[256,5000] = sigmoid( H2 @ fc1_w + fc1_b )   (K = 1024)
// N = 5000 -> 79 column blocks of 64 (last partial: zero-fill B, guard store)
// ---------------------------------------------------------------------------
#define NBLK_FC1 79   // ceil(5000/64)
__global__ __launch_bounds__(128)
void fc1_gemm(const float* __restrict__ A, const float* __restrict__ W,
              const float* __restrict__ bias, float* __restrict__ out) {
    __shared__ __align__(16) float  Afp[2][16 * 32];
    __shared__ __align__(16) __bf16 Bsh[2][64 * LDSTR];

    int tid  = threadIdx.x;
    int wv   = tid >> 5;
    int lane = tid & 31;
    int mn = lane & 15, half = lane >> 4;
    int m0 = (blockIdx.x / NBLK_FC1) * 16;
    int n0 = (blockIdx.x % NBLK_FC1) * 64;

    auto fill = [&](int buf, int k0) {
        int row = tid >> 3;
        int seg = (tid & 7) * 4;
        async_copy16(&Afp[buf][row * 32 + seg],
                     &A[(size_t)(m0 + row) * HID + k0 + seg]);
        for (int i = tid; i < 1024; i += 128) {
            int c = i & 63;
            int r = (i >> 6) * 2;
            bool ok = (n0 + c) < NCLS;
            float b0 = ok ? W[(size_t)(k0 + r) * NCLS + n0 + c] : 0.0f;
            float b1 = ok ? W[(size_t)(k0 + r + 1) * NCLS + n0 + c] : 0.0f;
            *(v2bf*)&Bsh[buf][c * LDSTR + r] = (v2bf){(__bf16)b0, (__bf16)b1};
        }
    };

    fill(0, 0);
    wait_async0();
    __syncthreads();
    v8f acc = {};
    for (int k0 = 0; k0 < HID; k0 += 32) {
        int cur = (k0 >> 5) & 1, nxt = cur ^ 1;
        v16bf a = frag_f32(&Afp[cur][mn * 32], half);
        v16bf b = frag_bf16(&Bsh[cur][(wv * 16 + mn) * LDSTR], half);
        if (k0 + 32 < HID) {
            __builtin_prefetch(&W[(size_t)(k0 + 64) * NCLS + n0 + lane * 2], 0, 1);
            fill(nxt, k0 + 32);
        }
        acc = __builtin_amdgcn_wmma_f32_16x16x32_bf16(false, a, false, b,
                                                      (short)0, acc, false, false);
        wait_async0();
        __syncthreads();
    }
#pragma unroll
    for (int r = 0; r < 8; ++r) {
        int row = m0 + half * 8 + r;
        int col = n0 + wv * 16 + mn;
        if (col < NCLS) {
            float v = acc[r] + bias[col];
            out[(size_t)row * NCLS + col] = 1.0f / (1.0f + __expf(-v));
        }
    }
}

// ---------------------------------------------------------------------------
// Launch
// ---------------------------------------------------------------------------
extern "C" void kernel_launch(void* const* d_in, const int* in_sizes, int n_in,
                              void* d_out, int out_size, void* d_ws, size_t ws_size,
                              hipStream_t stream) {
    const float* x        = (const float*)d_in[0];
    const int*   go_src   = (const int*)d_in[1];
    const int*   go_dst   = (const int*)d_in[2];
    const int*   go_et    = (const int*)d_in[3];
    const int*   blk_src  = (const int*)d_in[4];
    const int*   blk_dst  = (const int*)d_in[5];
    const float* w_rel    = (const float*)d_in[6];
    const float* w_loop   = (const float*)d_in[7];
    const float* b_rel    = (const float*)d_in[8];
    const float* fc0_w    = (const float*)d_in[9];
    const float* fc0_b    = (const float*)d_in[10];
    const float* gcn_w    = (const float*)d_in[11];
    const float* gcn_b    = (const float*)d_in[12];
    const float* fc1_w    = (const float*)d_in[13];
    const float* fc1_b    = (const float*)d_in[14];
    float* out = (float*)d_out;

    char* w = (char*)d_ws;
    float* agg    = (float*)w;                                    // NNODE
    float* H1     = (float*)(w + (size_t)NNODE * 4);              // 256*1024
    float* msum   = (float*)(w + (size_t)NNODE * 4 + 1048576);    // 256*1024
    float* H2     = (float*)(w + (size_t)NNODE * 4 + 2097152);    // 256*1024
    float* degout = (float*)(w + (size_t)NNODE * 4 + 3145728);    // 256
    float* degin  = degout + 256;

    // 1. RelGraphConv scatter
    zero_f32<<<(NNODE + 255) / 256, 256, 0, stream>>>(agg, NNODE);
    rel_scatter<<<(EGO + 255) / 256, 256, 0, stream>>>(go_src, go_dst, go_et,
                                                       x, w_rel, agg);
    // 2. fc0 + relu: 16 M-tiles x 16 N-blocks = 256 blocks
    fc0_gemm<<<256, 128, 0, stream>>>(agg, x, w_loop, b_rel, fc0_w, fc0_b, H1);

    // 3. GraphConv aggregation
    zero_f32<<<(NPROT * HID + 255) / 256, 256, 0, stream>>>(msum, NPROT * HID);
    zero_f32<<<2, 256, 0, stream>>>(degout, 512);
    blk_degrees<<<(EBLK + 255) / 256, 256, 0, stream>>>(blk_src, blk_dst, degout, degin);
    blk_scatter<<<EBLK, 256, 0, stream>>>(blk_src, blk_dst, H1, degout, msum);
    blk_norm_in<<<(NPROT * HID + 255) / 256, 256, 0, stream>>>(msum, degin);

    // 4. gcn GEMM
    gcn_gemm<<<256, 128, 0, stream>>>(msum, gcn_w, gcn_b, H2);

    // 5. fc1 + sigmoid: 16 x 79 = 1264 blocks
    fc1_gemm<<<16 * NBLK_FC1, 128, 0, stream>>>(H2, fc1_w, fc1_b, out);
}
